// EATN_26508538151246
// MI455X (gfx1250) — matrix operand order repaired and checked by reading, not compiled, not run on hardware
//
#include <hip/hip_runtime.h>
#include <math.h>

// ---------------------------------------------------------------------------
// CDNA5 / gfx1250 WMMA implementation of the EATN-style block.
// All GEMM-shaped work (3x3 convs via implicit im2col, 1x1 convs, MLP,
// N=1024 self-attention) runs on v_wmma_f32_16x16x32_bf16 (f32 accumulate).
// Each wave32 computes a register-blocked 16(M) x 64(N) tile: one A fragment
// feeds 4 WMMAs per K-chunk (4 accumulator sets). Branch-free K chunks,
// masked tail, float4 (global_load_b128) fragment feeds where K-stride == 1.
// ---------------------------------------------------------------------------

typedef __attribute__((ext_vector_type(16))) __bf16 v16bf;
typedef __attribute__((ext_vector_type(8)))  float  v8f;

#define BATCH 32
#define NPIX  1024   // 32*32
#define CDIM  128

struct GemmEpi {
  float scale;        // acc *= scale
  const float* b1;    // acc += b1[m]
  const float* g;     // acc *= g[m]
  const float* b2;    // acc += b2[m]
  int act;            // 0 none, 1 relu, 2 exact gelu
  const float* res;   // acc += res(b,m,n)
  long r_ms, r_ns, r_bs;
};

__device__ __forceinline__ v16bf pack16(const float* p, long s) {
  v16bf f;
#pragma unroll
  for (int j = 0; j < 16; ++j) f[j] = (__bf16)p[(long)j * s];
  return f;
}

__device__ __forceinline__ v16bf pack16_f4(const float* p) {
  float4 y0 = *(const float4*)(p);
  float4 y1 = *(const float4*)(p + 4);
  float4 y2 = *(const float4*)(p + 8);
  float4 y3 = *(const float4*)(p + 12);
  v16bf f;
  f[0]=(__bf16)y0.x;  f[1]=(__bf16)y0.y;  f[2]=(__bf16)y0.z;  f[3]=(__bf16)y0.w;
  f[4]=(__bf16)y1.x;  f[5]=(__bf16)y1.y;  f[6]=(__bf16)y1.z;  f[7]=(__bf16)y1.w;
  f[8]=(__bf16)y2.x;  f[9]=(__bf16)y2.y;  f[10]=(__bf16)y2.z; f[11]=(__bf16)y2.w;
  f[12]=(__bf16)y3.x; f[13]=(__bf16)y3.y; f[14]=(__bf16)y3.z; f[15]=(__bf16)y3.w;
  return f;
}

// ---------------------------------------------------------------------------
// Generic strided GEMM: C(b,m,n) = epi( scale * sum_k A(b,m,k) * B(b,k,n) )
// A (m,k) at A[b*a_bs + m*a_rs + k*a_cs]; B (k,n) at Bm[b*b_bs + k*b_ks + n*b_ns].
// ACONT: a_cs==1 (float4 loads). BCONT: b_ks==1. Nn must be a multiple of 64.
// ---------------------------------------------------------------------------
template<int ACONT, int BCONT>
__global__ __launch_bounds__(32)
void gemm_bf16_wmma(const float* __restrict__ A, long a_rs, long a_cs, long a_bs,
                    const float* __restrict__ Bm, long b_ks, long b_ns, long b_bs,
                    float* Cm, long c_ms, long c_ns, long c_bs,
                    int M, int Nn, int K, GemmEpi e)
{
  const int lane   = threadIdx.x;
  const int tilesN = Nn >> 6;            // 64-wide N tiles
  const int tm = blockIdx.x / tilesN;
  const int tn = blockIdx.x - tm * tilesN;
  const int b  = blockIdx.y;
  A  += (long)b * a_bs;
  Bm += (long)b * b_bs;
  Cm += (long)b * c_bs;

  const int l15   = lane & 15;
  const int hi    = lane >> 4;           // 0 or 1
  const int mrow  = tm * 16 + l15;
  const int ncol0 = tn * 64 + l15;       // + 16*u, u=0..3

  const float* pa = A + (long)mrow * a_rs + (long)(hi * 16) * a_cs;
  const float* pb[4];
#pragma unroll
  for (int u = 0; u < 4; ++u)
    pb[u] = Bm + (long)(ncol0 + 16 * u) * b_ns + (long)(hi * 16) * b_ks;

  const int KF = K & ~31;                // full chunks
  v8f acc[4] = {};
  for (int k0 = 0; k0 < KF; k0 += 32) {
    v16bf af;
    if (ACONT) {
      float4 x0 = *(const float4*)(pa);
      float4 x1 = *(const float4*)(pa + 4);
      float4 x2 = *(const float4*)(pa + 8);
      float4 x3 = *(const float4*)(pa + 12);
      af[0]=(__bf16)x0.x;  af[1]=(__bf16)x0.y;  af[2]=(__bf16)x0.z;  af[3]=(__bf16)x0.w;
      af[4]=(__bf16)x1.x;  af[5]=(__bf16)x1.y;  af[6]=(__bf16)x1.z;  af[7]=(__bf16)x1.w;
      af[8]=(__bf16)x2.x;  af[9]=(__bf16)x2.y;  af[10]=(__bf16)x2.z; af[11]=(__bf16)x2.w;
      af[12]=(__bf16)x3.x; af[13]=(__bf16)x3.y; af[14]=(__bf16)x3.z; af[15]=(__bf16)x3.w;
    } else {
#pragma unroll
      for (int j = 0; j < 8; ++j)  af[j]     = (__bf16)pa[(long)j * a_cs];
#pragma unroll
      for (int j = 0; j < 8; ++j)  af[8 + j] = (__bf16)pa[(long)(8 + j) * a_cs];
    }
    v16bf bfv[4];
#pragma unroll
    for (int u = 0; u < 4; ++u)
      bfv[u] = BCONT ? pack16_f4(pb[u]) : pack16(pb[u], b_ks);
#pragma unroll
    for (int u = 0; u < 4; ++u)
      acc[u] = __builtin_amdgcn_wmma_f32_16x16x32_bf16(false, af, false, bfv[u],
                                                       (short)0, acc[u], false, false);
    pa += 32 * a_cs;
#pragma unroll
    for (int u = 0; u < 4; ++u) pb[u] += 32 * b_ks;
  }

  if (KF < K) {                          // masked tail chunk (branch-free)
    v16bf af;
#pragma unroll
    for (int j = 0; j < 16; ++j) {
      int ko = hi * 16 + j;
      int k  = KF + ko;
      float m = (k < K) ? 1.0f : 0.0f;
      int  kc = (k < K) ? k : (K - 1);
      af[j] = (__bf16)(m * A[(long)mrow * a_rs + (long)kc * a_cs]);
    }
#pragma unroll
    for (int u = 0; u < 4; ++u) {
      v16bf bfv;
#pragma unroll
      for (int j = 0; j < 16; ++j) {
        int k = KF + hi * 16 + j;
        float m = (k < K) ? 1.0f : 0.0f;
        int  kc = (k < K) ? k : (K - 1);
        bfv[j] = (__bf16)(m * Bm[(long)kc * b_ks + (long)(ncol0 + 16 * u) * b_ns]);
      }
      acc[u] = __builtin_amdgcn_wmma_f32_16x16x32_bf16(false, af, false, bfv,
                                                       (short)0, acc[u], false, false);
    }
  }

  const int mbase = tm * 16 + hi * 8;
#pragma unroll
  for (int u = 0; u < 4; ++u) {
    const int nc = ncol0 + 16 * u;
#pragma unroll
    for (int r = 0; r < 8; ++r) {
      int m = mbase + r;
      float v = acc[u][r] * e.scale;
      if (e.b1) v += e.b1[m];
      if (e.g)  v *= e.g[m];
      if (e.b2) v += e.b2[m];
      if (e.act == 1)      v = fmaxf(v, 0.0f);
      else if (e.act == 2) v = 0.5f * v * (1.0f + erff(v * 0.70710678118654752440f));
      if (e.res) v += e.res[(long)blockIdx.y * e.r_bs + (long)m * e.r_ms + (long)nc * e.r_ns];
      Cm[(long)m * c_ms + (long)nc * c_ns] = v;
    }
  }
}

// ---------------------------------------------------------------------------
// 3x3 SAME conv as implicit-im2col WMMA GEMM, tap-outer so the spatial halo
// predicate is a hoisted per-tap multiply mask (branch-free inner chunks).
// Each wave computes 16 couts x 64 pixels; weight fragment reused 4x.
//   Y(b,co,n) = relu( g[co]*conv + bia[co] ) (+ res(b,co,n))
// ---------------------------------------------------------------------------
template<int CIN>
__global__ __launch_bounds__(32)
void conv3x3_bf16_wmma(const float* __restrict__ X, const float* __restrict__ Wt,
                       const float* __restrict__ g, const float* __restrict__ bia,
                       const float* res, float* Y, int Cout)
{
  const int lane = threadIdx.x;
  const int tm = blockIdx.x >> 4;        // Cout/16 tiles
  const int tn = blockIdx.x & 15;        // 16 pixel tiles of 64
  const int b  = blockIdx.y;
  const float* Xb = X + (long)b * CIN * NPIX;

  const int K     = CIN * 9;
  const int l15   = lane & 15;
  const int hi    = lane >> 4;
  const int mrow  = tm * 16 + l15;
  const int ncol0 = tn * 64 + l15;       // + 16*u

  v8f acc[4] = {};
  for (int tap = 0; tap < 9; ++tap) {
    const int ky = tap / 3, kx = tap - ky * 3;
    float bm[4];
    const float* xb[4];
#pragma unroll
    for (int u = 0; u < 4; ++u) {
      const int nc = ncol0 + 16 * u;
      const int iy = (nc >> 5) + ky - 1, ix = (nc & 31) + kx - 1;
      const bool vld = (iy >= 0) && (iy < 32) && (ix >= 0) && (ix < 32);
      bm[u] = vld ? 1.0f : 0.0f;
      xb[u] = Xb + (vld ? (iy * 32 + ix) : 0);
    }
    const float* wb = Wt + (long)mrow * K + tap;   // + ci*9 per element
    for (int c0 = 0; c0 < CIN; c0 += 32) {
      v16bf af;
#pragma unroll
      for (int j = 0; j < 16; ++j) {
        int ci = c0 + hi * 16 + j;
        float m = (ci < CIN) ? 1.0f : 0.0f;
        int  cc = (ci < CIN) ? ci : (CIN - 1);
        af[j] = (__bf16)(m * wb[cc * 9]);
      }
#pragma unroll
      for (int u = 0; u < 4; ++u) {
        v16bf bfv;
#pragma unroll
        for (int j = 0; j < 16; ++j) {
          int ci = c0 + hi * 16 + j;
          float m = (ci < CIN) ? bm[u] : 0.0f;
          int  cc = (ci < CIN) ? ci : (CIN - 1);
          bfv[j] = (__bf16)(m * xb[u][cc * NPIX]);
        }
        acc[u] = __builtin_amdgcn_wmma_f32_16x16x32_bf16(false, af, false, bfv,
                                                         (short)0, acc[u], false, false);
      }
    }
  }

  const int mbase = tm * 16 + hi * 8;
#pragma unroll
  for (int u = 0; u < 4; ++u) {
    const int nc = ncol0 + 16 * u;
#pragma unroll
    for (int r = 0; r < 8; ++r) {
      int m = mbase + r;
      float v = fmaxf(acc[u][r] * g[m] + bia[m], 0.0f);
      long o = ((long)b * Cout + m) * NPIX + nc;
      if (res) v += res[o];
      Y[o] = v;
    }
  }
}

// ---------------------------------------------------------------------------
// LayerNorm over 128 channels, channel-major (B,128,1024). One thread / token
// (consecutive threads -> consecutive n -> coalesced per-channel rows).
// ---------------------------------------------------------------------------
__global__ void ln_kernel(const float* __restrict__ T, const float* __restrict__ g,
                          const float* __restrict__ b, float* __restrict__ O)
{
  long idx = (long)blockIdx.x * blockDim.x + threadIdx.x;
  if (idx >= (long)BATCH * NPIX) return;
  int bb = (int)(idx >> 10), n = (int)(idx & 1023);
  const float* p = T + ((long)bb * CDIM) * NPIX + n;
  float mu = 0.f;
  for (int c = 0; c < CDIM; ++c) mu += p[c * NPIX];
  mu *= (1.0f / CDIM);
  float va = 0.f;
  for (int c = 0; c < CDIM; ++c) { float d = p[c * NPIX] - mu; va += d * d; }
  va *= (1.0f / CDIM);
  float inv = rsqrtf(va + 1e-5f);
  float* o = O + ((long)bb * CDIM) * NPIX + n;
  for (int c = 0; c < CDIM; ++c) o[c * NPIX] = (p[c * NPIX] - mu) * inv * g[c] + b[c];
}

// ----- GISSA support kernels (tiny FLOP count; VALU) -----------------------
__global__ void qkv1_kernel(const float* __restrict__ cur, const float* __restrict__ w,
                            const float* __restrict__ bias, float* __restrict__ qkv)
{
  long idx = (long)blockIdx.x * blockDim.x + threadIdx.x;
  if (idx >= (long)BATCH * 384 * NPIX) return;
  int n = (int)(idx & 1023);
  int r = (int)((idx >> 10) % 384);
  int b = (int)(idx / (384L * NPIX));
  int grp = r / 24;                        // groups=16, 8 in / 24 out
  const float* xin = cur + ((long)b * CDIM + grp * 8) * NPIX + n;
  const float* wr  = w + r * 8;
  float s = bias[r];
#pragma unroll
  for (int c = 0; c < 8; ++c) s += xin[c * NPIX] * wr[c];
  qkv[idx] = s;                            // layout (b, r=s*128+h*8+d, n)
}

__global__ void attn1_kernel(const float* __restrict__ qkv, float* __restrict__ attn1)
{
  int bh = blockIdx.x;                     // b*16+h
  int b = bh >> 4, h = bh & 15;
  int tid = threadIdx.x;                   // 64
  int d = tid >> 3, e = tid & 7;
  const float4* q = (const float4*)(qkv + ((long)b * 384 + h * 8 + d) * NPIX);
  const float4* k = (const float4*)(qkv + ((long)b * 384 + 128 + h * 8 + e) * NPIX);
  float s = 0.f;
  for (int n = 0; n < 256; ++n) {
    float4 qa = q[n], ka = k[n];
    s += qa.x * ka.x + qa.y * ka.y + qa.z * ka.z + qa.w * ka.w;
  }
  s *= 0.35355339059327373f;               // 8^-0.5
  float a = ((s >= 0.f) ? 1.f : -1.f) * sqrtf(fabsf(s) + 1e-5f);
  __shared__ float L[64];
  L[tid] = a;
  __syncthreads();
  if (tid < 8) {
    float mx = -1e30f;
    for (int j = 0; j < 8; ++j) mx = fmaxf(mx, L[tid * 8 + j]);
    float sm = 0.f, ex[8];
    for (int j = 0; j < 8; ++j) { ex[j] = expf(L[tid * 8 + j] - mx); sm += ex[j]; }
    for (int j = 0; j < 8; ++j) attn1[((long)bh * 8 + tid) * 8 + j] = ex[j] / sm;
  }
}

// y[b,c',n] = (sum_e attn1*v + cur[b,c,n]) * g[c'] + b[c'],  c'=d*16+h, c=h*8+d
__global__ void av1_kernel(const float* __restrict__ qkv, const float* __restrict__ attn1,
                           const float* __restrict__ cur, const float* __restrict__ gg,
                           const float* __restrict__ gb, float* __restrict__ y)
{
  long idx = (long)blockIdx.x * blockDim.x + threadIdx.x;
  if (idx >= (long)BATCH * CDIM * NPIX) return;
  int n  = (int)(idx & 1023);
  int cp = (int)((idx >> 10) & 127);
  int b  = (int)(idx / (128L * NPIX));
  int d = cp >> 4, h = cp & 15;
  const float* a = attn1 + (((long)b * 16 + h) * 8 + d) * 8;
  const float* v = qkv + ((long)b * 384 + 256 + h * 8) * NPIX + n;
  float s = 0.f;
#pragma unroll
  for (int e2 = 0; e2 < 8; ++e2) s += a[e2] * v[e2 * NPIX];
  s += cur[((long)b * CDIM + h * 8 + d) * NPIX + n];
  y[idx] = s * gg[cp] + gb[cp];
}

__global__ void qkv2_kernel(const float* __restrict__ y, const float* __restrict__ w,
                            const float* __restrict__ bias, float* __restrict__ qkv2)
{
  long idx = (long)blockIdx.x * blockDim.x + threadIdx.x;
  if (idx >= (long)BATCH * 384 * NPIX) return;
  int n = (int)(idx & 1023);
  int r = (int)((idx >> 10) % 384);
  int b = (int)(idx / (384L * NPIX));
  int grp = r / 48;                        // groups=8, 16 in / 48 out
  const float* xin = y + ((long)b * CDIM + grp * 16) * NPIX + n;
  const float* wr  = w + r * 16;
  float s = bias[r];
#pragma unroll
  for (int c = 0; c < 16; ++c) s += fmaxf(xin[c * NPIX], 0.0f) * wr[c];  // xr = relu(y)
  qkv2[idx] = s;                           // (b, r=s*128+d*16+h, n)
}

__global__ void attn2_kernel(const float* __restrict__ qkv2, float* __restrict__ attn2)
{
  int bd = blockIdx.x;                     // b*8+d
  int b = bd >> 3, d = bd & 7;
  int tid = threadIdx.x;                   // 256
  int h = tid >> 4, g2 = tid & 15;
  const float4* q = (const float4*)(qkv2 + ((long)b * 384 + d * 16 + h) * NPIX);
  const float4* k = (const float4*)(qkv2 + ((long)b * 384 + 128 + d * 16 + g2) * NPIX);
  float s = 0.f;
  for (int n = 0; n < 256; ++n) {
    float4 qa = q[n], ka = k[n];
    s += qa.x * ka.x + qa.y * ka.y + qa.z * ka.z + qa.w * ka.w;
  }
  s *= 0.25f;                              // 16^-0.5
  float a = ((s >= 0.f) ? 1.f : -1.f) * sqrtf(fabsf(s) + 1e-5f);
  __shared__ float L[256];
  L[tid] = a;
  __syncthreads();
  if (tid < 16) {
    float mx = -1e30f;
    for (int j = 0; j < 16; ++j) mx = fmaxf(mx, L[tid * 16 + j]);
    float sm = 0.f, ex[16];
    for (int j = 0; j < 16; ++j) { ex[j] = expf(L[tid * 16 + j] - mx); sm += ex[j]; }
    for (int j = 0; j < 16; ++j) attn2[((long)bd * 16 + tid) * 16 + j] = ex[j] / sm;
  }
}

// t[b,c,n] += (sum_g attn2[b,d,h,g]*v2[b,d,g,n]) + y[b,c,n],  c=h*8+d
__global__ void av2_kernel(const float* __restrict__ qkv2, const float* __restrict__ attn2,
                           const float* __restrict__ y, float* __restrict__ t)
{
  long idx = (long)blockIdx.x * blockDim.x + threadIdx.x;
  if (idx >= (long)BATCH * CDIM * NPIX) return;
  int n = (int)(idx & 1023);
  int c = (int)((idx >> 10) & 127);
  int b = (int)(idx / (128L * NPIX));
  int h = c >> 3, d = c & 7;
  const float* a = attn2 + (((long)b * 8 + d) * 16 + h) * 16;
  const float* v = qkv2 + ((long)b * 384 + 256 + d * 16) * NPIX + n;
  float s = 0.f;
#pragma unroll
  for (int g2 = 0; g2 < 16; ++g2) s += a[g2] * v[g2 * NPIX];
  t[idx] += s + y[idx];
}

// ---------------------------------------------------------------------------
// Row softmax over 1024 elements. One block (256 threads) per row; each
// thread owns exactly one float4 of the row.
// ---------------------------------------------------------------------------
__global__ __launch_bounds__(256)
void softmax1024_kernel(float* __restrict__ A)
{
  float4* p = (float4*)(A + (long)blockIdx.x * 1024);
  int tid = threadIdx.x;
  __shared__ float red[256];
  float4 v = p[tid];
  float mx = fmaxf(fmaxf(v.x, v.y), fmaxf(v.z, v.w));
  red[tid] = mx; __syncthreads();
  for (int s = 128; s > 0; s >>= 1) { if (tid < s) red[tid] = fmaxf(red[tid], red[tid + s]); __syncthreads(); }
  mx = red[0]; __syncthreads();
  v.x = expf(v.x - mx); v.y = expf(v.y - mx); v.z = expf(v.z - mx); v.w = expf(v.w - mx);
  red[tid] = v.x + v.y + v.z + v.w; __syncthreads();
  for (int s = 128; s > 0; s >>= 1) { if (tid < s) red[tid] += red[tid + s]; __syncthreads(); }
  float inv = 1.0f / red[0];
  v.x *= inv; v.y *= inv; v.z *= inv; v.w *= inv;
  p[tid] = v;
}

// ---------------------------------------------------------------------------
// Head: out[b,o] = fc_w[o,:] . mean_n( lmd*xc + (1-lmd)*xp )
// ---------------------------------------------------------------------------
__global__ __launch_bounds__(128)
void head_kernel(const float* __restrict__ xc, const float* __restrict__ xp,
                 const float* __restrict__ lamuda, const float* __restrict__ fcw,
                 float* __restrict__ out)
{
  int b = blockIdx.x, tid = threadIdx.x;   // 128 threads
  float lmd = 1.0f / (1.0f + expf(-lamuda[0]));
  const float4* pc = (const float4*)(xc + ((long)b * CDIM + tid) * NPIX);
  const float4* pp = (const float4*)(xp + ((long)b * CDIM + tid) * NPIX);
  float s = 0.f;
  for (int n = 0; n < 256; ++n) {
    float4 a = pc[n], c = pp[n];
    s += lmd * (a.x + a.y + a.z + a.w) + (1.0f - lmd) * (c.x + c.y + c.z + c.w);
  }
  __shared__ float pool[CDIM];
  pool[tid] = s * (1.0f / NPIX);
  __syncthreads();
  if (tid < 16) {
    float o = 0.f;
    for (int c = 0; c < CDIM; ++c) o += pool[c] * fcw[tid * CDIM + c];
    out[b * 16 + tid] = o;
  }
}

// ---------------------------------------------------------------------------
static inline void launch_gemm(hipStream_t st,
    const float* A, long a_rs, long a_cs, long a_bs,
    const float* Bm, long b_ks, long b_ns, long b_bs,
    float* Cm, long c_ms, long c_ns, long c_bs,
    int M, int Nn, int K, int batch, GemmEpi e)
{
  dim3 grid((M / 16) * (Nn / 64), batch);
  const bool ac = (a_cs == 1), bc = (b_ks == 1);
  if (ac && bc)
    gemm_bf16_wmma<1,1><<<grid, 32, 0, st>>>(A,a_rs,a_cs,a_bs, Bm,b_ks,b_ns,b_bs, Cm,c_ms,c_ns,c_bs, M,Nn,K, e);
  else if (ac)
    gemm_bf16_wmma<1,0><<<grid, 32, 0, st>>>(A,a_rs,a_cs,a_bs, Bm,b_ks,b_ns,b_bs, Cm,c_ms,c_ns,c_bs, M,Nn,K, e);
  else if (bc)
    gemm_bf16_wmma<0,1><<<grid, 32, 0, st>>>(A,a_rs,a_cs,a_bs, Bm,b_ks,b_ns,b_bs, Cm,c_ms,c_ns,c_bs, M,Nn,K, e);
  else
    gemm_bf16_wmma<0,0><<<grid, 32, 0, st>>>(A,a_rs,a_cs,a_bs, Bm,b_ks,b_ns,b_bs, Cm,c_ms,c_ns,c_bs, M,Nn,K, e);
}

extern "C" void kernel_launch(void* const* d_in, const int* in_sizes, int n_in,
                              void* d_out, int out_size, void* d_ws, size_t ws_size,
                              hipStream_t stream)
{
  const float* x      = (const float*)d_in[0];
  const float* ssfe_w = (const float*)d_in[1];
  const float* ssfe_g = (const float*)d_in[2];
  const float* ssfe_b = (const float*)d_in[3];
  const float* cc_w   = (const float*)d_in[4];
  const float* cc_g   = (const float*)d_in[5];
  const float* cc_b   = (const float*)d_in[6];
  const float* ln1_g  = (const float*)d_in[7];
  const float* ln1_b  = (const float*)d_in[8];
  const float* qkv_w  = (const float*)d_in[9];
  const float* qkv_b  = (const float*)d_in[10];
  const float* gbn_g  = (const float*)d_in[11];
  const float* gbn_b  = (const float*)d_in[12];
  const float* qkv2_w = (const float*)d_in[13];
  const float* qkv2_b = (const float*)d_in[14];
  const float* ln2_g  = (const float*)d_in[15];
  const float* ln2_b  = (const float*)d_in[16];
  const float* mlp_w1 = (const float*)d_in[17];
  const float* mlp_b1 = (const float*)d_in[18];
  const float* mlp_w2 = (const float*)d_in[19];
  const float* mlp_b2 = (const float*)d_in[20];
  const float* cs_w   = (const float*)d_in[21];
  const float* cs_g   = (const float*)d_in[22];
  const float* cs_b   = (const float*)d_in[23];
  const float* lfe0_w = (const float*)d_in[24];
  const float* lfe0_g = (const float*)d_in[25];
  const float* lfe0_b = (const float*)d_in[26];
  const float* s0_pi_w = (const float*)d_in[27];
  const float* s0_pi_b = (const float*)d_in[28];
  const float* s0_bn_g = (const float*)d_in[29];
  const float* s0_bn_b = (const float*)d_in[30];
  const float* s0_po_w = (const float*)d_in[31];
  const float* s0_po_b = (const float*)d_in[32];
  const float* lfe1_w = (const float*)d_in[33];
  const float* lfe1_g = (const float*)d_in[34];
  const float* lfe1_b = (const float*)d_in[35];
  const float* s1_pi_w = (const float*)d_in[36];
  const float* s1_pi_b = (const float*)d_in[37];
  const float* s1_bn_g = (const float*)d_in[38];
  const float* s1_bn_b = (const float*)d_in[39];
  const float* s1_po_w = (const float*)d_in[40];
  const float* s1_po_b = (const float*)d_in[41];
  const float* lamuda  = (const float*)d_in[42];
  const float* fc_w    = (const float*)d_in[43];
  float* out = (float*)d_out;

  // ---- workspace layout (floats) ----
  float* W = (float*)d_ws;
  float* xs   = W;                         // 32*144*1024
  float* t    = xs  + 4718592;             // 32*128*1024 (token stream, channel-major)
  float* cur  = t   + 4194304;             // LN output / scratch
  float* big  = cur + 4194304;             // 32*512*1024 (qkv1/qkv2/mlp hidden)
  float* yg   = big + 16777216;            // GISSA y (pre-relu)
  float* a1   = yg  + 4194304;             // attn1 32*16*8*8
  float* a2   = a1  + 32768;               // attn2 32*8*16*16
  float* xp   = a2  + 65536;               // 32*128*1024
  float* xp2  = xp  + 4194304;             // 32*128*1024
  float* pt   = xp2 + 4194304;             // 32*256*1024 (q|v)
  float* atn  = pt  + 8388608;             // 32*1024*1024 attention map
  float* yb   = atn + 33554432;            // 32*128*1024 attn@v result
  (void)ws_size; (void)in_sizes; (void)n_in; (void)out_size;

  GemmEpi E0 = {1.f, nullptr, nullptr, nullptr, 0, nullptr, 0, 0, 0};
  const long CB = (long)CDIM * NPIX;       // per-batch 128*1024

  // 1) xs = cbr3(x, ssfe)         [WMMA, K=1296]
  conv3x3_bf16_wmma<144><<<dim3(9 * 16, BATCH), 32, 0, stream>>>(x, ssfe_w, ssfe_g, ssfe_b, nullptr, xs, 144);

  // 2) t = cbr1(xs, cc)           [WMMA GEMM 128x1024x144]
  { GemmEpi e = E0; e.g = cc_g; e.b2 = cc_b; e.act = 1;
    launch_gemm(stream, cc_w, 144, 1, 0, xs, NPIX, 1, 144L * NPIX,
                t, NPIX, 1, CB, 128, NPIX, 144, BATCH, e); }

  // 3) cur = LN1(t)
  ln_kernel<<<(BATCH * NPIX + 255) / 256, 256, 0, stream>>>(t, ln1_g, ln1_b, cur);

  // 4-9) GISSA (small grouped attention, VALU)
  { long tot = (long)BATCH * 384 * NPIX;
    qkv1_kernel<<<(tot + 255) / 256, 256, 0, stream>>>(cur, qkv_w, qkv_b, big); }
  attn1_kernel<<<BATCH * 16, 64, 0, stream>>>(big, a1);
  { long tot = (long)BATCH * CDIM * NPIX;
    av1_kernel<<<(tot + 255) / 256, 256, 0, stream>>>(big, a1, cur, gbn_g, gbn_b, yg); }
  { long tot = (long)BATCH * 384 * NPIX;
    qkv2_kernel<<<(tot + 255) / 256, 256, 0, stream>>>(yg, qkv2_w, qkv2_b, big); }
  attn2_kernel<<<BATCH * 8, 256, 0, stream>>>(big, a2);
  { long tot = (long)BATCH * CDIM * NPIX;
    av2_kernel<<<(tot + 255) / 256, 256, 0, stream>>>(big, a2, yg, t); }

  // 10) cur = LN2(t)
  ln_kernel<<<(BATCH * NPIX + 255) / 256, 256, 0, stream>>>(t, ln2_g, ln2_b, cur);

  // 11) big = gelu(W1 @ cur + b1) [WMMA GEMM 512x1024x128]
  { GemmEpi e = E0; e.b1 = mlp_b1; e.act = 2;
    launch_gemm(stream, mlp_w1, 128, 1, 0, cur, NPIX, 1, CB,
                big, NPIX, 1, 512L * NPIX, 512, NPIX, 128, BATCH, e); }

  // 12) t += W2 @ big + b2        [WMMA GEMM 128x1024x512]
  { GemmEpi e = E0; e.b1 = mlp_b2; e.res = t; e.r_ms = NPIX; e.r_ns = 1; e.r_bs = CB;
    launch_gemm(stream, mlp_w2, 512, 1, 0, big, NPIX, 1, 512L * NPIX,
                t, NPIX, 1, CB, 128, NPIX, 512, BATCH, e); }

  // 13) xp = cbr3(xs, cs)         [WMMA, K=1296]
  conv3x3_bf16_wmma<144><<<dim3(8 * 16, BATCH), 32, 0, stream>>>(xs, cs_w, cs_g, cs_b, nullptr, xp, 128);

  // 14) xp2 = cbr3(xp, lfe0) + xp [WMMA, K=1152]
  conv3x3_bf16_wmma<128><<<dim3(8 * 16, BATCH), 32, 0, stream>>>(xp, lfe0_w, lfe0_g, lfe0_b, xp, xp2, 128);

  // 15) pt = (pi0(xp2)+b)*g+b2    [WMMA GEMM 256x1024x128]
  { GemmEpi e = E0; e.b1 = s0_pi_b; e.g = s0_bn_g; e.b2 = s0_bn_b;
    launch_gemm(stream, s0_pi_w, 128, 1, 0, xp2, NPIX, 1, CB,
                pt, NPIX, 1, 256L * NPIX, 256, NPIX, 128, BATCH, e); }

  // 16) atn = q^T q * 128^-0.5    [WMMA GEMM 1024x1024x128  -- the big one]
  { GemmEpi e = E0; e.scale = 0.08838834764831845f;
    launch_gemm(stream, pt, 1, NPIX, 256L * NPIX, pt, NPIX, 1, 256L * NPIX,
                atn, NPIX, 1, (long)NPIX * NPIX, NPIX, NPIX, 128, BATCH, e); }

  // 17) softmax rows of atn
  softmax1024_kernel<<<BATCH * NPIX, 256, 0, stream>>>(atn);

  // 18) yb = v @ atn^T            [WMMA GEMM 128x1024x1024, A & B both K-contiguous]
  launch_gemm(stream, pt + 128L * NPIX, NPIX, 1, 256L * NPIX,
              atn, 1, NPIX, (long)NPIX * NPIX,
              yb, NPIX, 1, CB, 128, NPIX, NPIX, BATCH, E0);

  // 19) xp = po0(yb)+b + xp2      [WMMA GEMM 128x1024x128]
  { GemmEpi e = E0; e.b1 = s0_po_b; e.res = xp2; e.r_ms = NPIX; e.r_ns = 1; e.r_bs = CB;
    launch_gemm(stream, s0_po_w, 128, 1, 0, yb, NPIX, 1, CB,
                xp, NPIX, 1, CB, 128, NPIX, 128, BATCH, e); }

  // 20) xp2 = cbr3(xp, lfe1) + xp [WMMA, K=1152]
  conv3x3_bf16_wmma<128><<<dim3(8 * 16, BATCH), 32, 0, stream>>>(xp, lfe1_w, lfe1_g, lfe1_b, xp, xp2, 128);

  // 21) pt = (pi1(xp2)+b)*g+b2    [WMMA GEMM 128x1024x128]  (v for reuse stage)
  { GemmEpi e = E0; e.b1 = s1_pi_b; e.g = s1_bn_g; e.b2 = s1_bn_b;
    launch_gemm(stream, s1_pi_w, 128, 1, 0, xp2, NPIX, 1, CB,
                pt, NPIX, 1, CB, 128, NPIX, 128, BATCH, e); }

  // 22) yb = (v @ atn^T)*128^-0.5 [WMMA GEMM 128x1024x1024]
  { GemmEpi e = E0; e.scale = 0.08838834764831845f;
    launch_gemm(stream, pt, NPIX, 1, CB, atn, 1, NPIX, (long)NPIX * NPIX,
                yb, NPIX, 1, CB, 128, NPIX, NPIX, BATCH, e); }

  // 23) xp = po1(yb)+b + xp2      [WMMA GEMM 128x1024x128]
  { GemmEpi e = E0; e.b1 = s1_po_b; e.res = xp2; e.r_ms = NPIX; e.r_ns = 1; e.r_bs = CB;
    launch_gemm(stream, s1_po_w, 128, 1, 0, yb, NPIX, 1, CB,
                xp, NPIX, 1, CB, 128, NPIX, 128, BATCH, e); }

  // 24) out = fc( mean( sigmoid(lmd)*t + (1-lmd)*xp ) )
  head_kernel<<<BATCH, 128, 0, stream>>>(t, xp, lamuda, fc_w, out);
}
